// AnomalyTransformerAE_69870527971490
// MI455X (gfx1250) — matrix operand order repaired and checked by reading, not compiled, not run on hardware
//
#include <hip/hip_runtime.h>
#include <stdint.h>

// Problem constants (match reference)
#define Bc  32
#define Lc  256
#define Cc  55
#define Dc  512
#define Hc  8
#define NLc 3
#define DFc 512
#define Zc  16
#define DKc 64

static const long long BLD   = (long long)Bc * Lc * Dc;        // 4194304
static const long long BHLL  = (long long)Bc * Hc * Lc * Lc;   // 16777216
static const long long DDsz  = (long long)Dc * Dc;             // 262144
static const long long RECON = (long long)Bc * Lc * Cc;        // 450560

typedef __attribute__((ext_vector_type(16))) __bf16 v16bf;
typedef __attribute__((ext_vector_type(8)))  float  v8f;

union FragB16 { uint4 u4[2]; v16bf v; };

__device__ __forceinline__ unsigned short f2bf(float f) {
  unsigned int u = __float_as_uint(f);
  u += 0x7fffu + ((u >> 16) & 1u);           // round-to-nearest-even
  return (unsigned short)(u >> 16);
}

// ---------------------------------------------------------------------------
// Generic bf16 WMMA GEMM:  out[m,n] = alpha * sum_k A[m,k]*B[n,k] + bias[n]
// A: M x K row-major bf16 (lda). B: if bKN==0, N x K row-major (weights, "Wt");
// if bKN==1, K x N row-major (e.g. V matrix) and is transposed into LDS.
// Batched over blockIdx.z: base(op) = (z/zDiv)*sOuter + (z%zDiv)*sInner.
// epilogue: 0 = none, 1 = exact GELU. Writes fp32 (outF) and/or bf16 (outB).
// Block tile 128x64x32; 8 waves; wave tile 32x32 = 2x2 WMMA 16x16x32 tiles.
// ---------------------------------------------------------------------------
__global__ __launch_bounds__(256) void gemm_bf16_wmma(
    const unsigned short* __restrict__ A, int lda,
    const unsigned short* __restrict__ Bm, int ldb, int bKN,
    const float* __restrict__ bias,
    float* __restrict__ outF, unsigned short* __restrict__ outB, int ldc,
    int K, float alpha, int epilogue, int zDiv,
    long long sAo, long long sAi, long long sBo, long long sBi,
    long long sCo, long long sCi)
{
  __shared__ unsigned short lsA[128 * 40];   // 128 rows x (32 + 8 pad) bf16
  __shared__ unsigned short lsB[64 * 40];    // 64 n-rows x (32 + 8 pad) bf16

  const int tid   = threadIdx.x;
  const int lane  = tid & 31;
  const int wid   = tid >> 5;
  const int warpM = wid & 3;                 // 4 waves along M
  const int warpN = wid >> 2;                // 2 waves along N
  const int m0 = blockIdx.x * 128;
  const int n0 = blockIdx.y * 64;

  const long long z  = blockIdx.z;
  const long long zo = z / zDiv, zi = z % zDiv;
  A  += zo * sAo + zi * sAi;
  Bm += zo * sBo + zi * sBi;
  const long long cbase = zo * sCo + zi * sCi;

  v8f acc[2][2];
  #pragma unroll
  for (int i = 0; i < 2; ++i)
    #pragma unroll
    for (int j = 0; j < 2; ++j)
      acc[i][j] = (v8f){0.f,0.f,0.f,0.f,0.f,0.f,0.f,0.f};

  const int aRow  = tid >> 1,  aHalf = (tid & 1) * 16;  // A staging
  const int bNr   = tid >> 2,  bCh   = (tid & 3) * 8;   // B staging (NK)
  const int bKr   = tid >> 3,  bNc   = (tid & 7) * 8;   // B staging (KN)
  const int khalf = lane >> 4;                          // ISA frag layouts
  const int r16   = lane & 15;

  for (int k0 = 0; k0 < K; k0 += 32) {
    __syncthreads();
    {
      const unsigned short* s = A + (long long)(m0 + aRow) * lda + k0 + aHalf;
      uint4 p0 = *(const uint4*)s;
      uint4 p1 = *(const uint4*)(s + 8);
      *(uint4*)&lsA[aRow * 40 + aHalf]     = p0;
      *(uint4*)&lsA[aRow * 40 + aHalf + 8] = p1;
      if (k0 + 32 < K) __builtin_prefetch(s + 32, 0, 0);   // global_prefetch_b8
    }
    if (!bKN) {
      uint4 p = *(const uint4*)(Bm + (long long)(n0 + bNr) * ldb + k0 + bCh);
      *(uint4*)&lsB[bNr * 40 + bCh] = p;
    } else {
      alignas(16) unsigned short t[8];
      *(uint4*)t = *(const uint4*)(Bm + (long long)(k0 + bKr) * ldb + n0 + bNc);
      #pragma unroll
      for (int e = 0; e < 8; ++e) lsB[(bNc + e) * 40 + bKr] = t[e];
    }
    __syncthreads();

    FragB16 af[2], bfr[2];
    #pragma unroll
    for (int tm = 0; tm < 2; ++tm) {          // A 16x32: lanes 0-15 K{0..7,16..23}
      int row = warpM * 32 + tm * 16 + r16;   //          lanes16-31 K{8..15,24..31}
      af[tm].u4[0] = *(const uint4*)&lsA[row * 40 + khalf * 8];
      af[tm].u4[1] = *(const uint4*)&lsA[row * 40 + 16 + khalf * 8];
    }
    #pragma unroll
    for (int tn = 0; tn < 2; ++tn) {          // B 32x16: lane=N, K{0..7,16..23} / K{8..15,24..31}
      int col = warpN * 32 + tn * 16 + r16;
      bfr[tn].u4[0] = *(const uint4*)&lsB[col * 40 + khalf * 8];
      bfr[tn].u4[1] = *(const uint4*)&lsB[col * 40 + 16 + khalf * 8];
    }
    #pragma unroll
    for (int tm = 0; tm < 2; ++tm)
      #pragma unroll
      for (int tn = 0; tn < 2; ++tn)
        acc[tm][tn] = __builtin_amdgcn_wmma_f32_16x16x32_bf16(
            false, af[tm].v, false, bfr[tn].v, (short)0, acc[tm][tn], false, false);
  }

  #pragma unroll
  for (int tm = 0; tm < 2; ++tm) {
    #pragma unroll
    for (int tn = 0; tn < 2; ++tn) {
      const int   ng = n0 + warpN * 32 + tn * 16 + r16;
      const float bv = bias ? bias[ng] : 0.0f;
      #pragma unroll
      for (int r = 0; r < 8; ++r) {           // C/D: VGPR r -> M=r+8*khalf, N=lane&15
        const int mg = m0 + warpM * 32 + tm * 16 + r + khalf * 8;
        float v = acc[tm][tn][r] * alpha + bv;
        if (epilogue == 1) v = 0.5f * v * (1.0f + erff(v * 0.70710678118f));
        const long long off = cbase + (long long)mg * ldc + ng;
        if (outF) outF[off] = v;
        if (outB) outB[off] = f2bf(v);
      }
    }
  }
}

// ---------------------------------------------------------------------------
__global__ void cvt_f32_bf16(const float* __restrict__ s,
                             unsigned short* __restrict__ d, int n) {
  int i = blockIdx.x * blockDim.x + threadIdx.x;
  if (i < n) d[i] = f2bf(s[i]);
}

// circular conv1d(C->D, k=3) + sinusoidal positional embedding
__global__ __launch_bounds__(256) void embed_kernel(
    const float* __restrict__ x, const float* __restrict__ tw,
    float* __restrict__ h, unsigned short* __restrict__ hb)
{
  int idx = blockIdx.x * 256 + threadIdx.x;        // (b,l,d)
  int d = idx & (Dc - 1);
  int l = (idx >> 9) & (Lc - 1);
  int b = idx >> 17;
  float s = 0.f;
  #pragma unroll
  for (int w = 0; w < 3; ++w) {
    int ls = l + w - 1;
    ls = (ls < 0) ? ls + Lc : (ls >= Lc ? ls - Lc : ls);
    const float* xr  = x  + ((long long)b * Lc + ls) * Cc;
    const float* twr = tw + (long long)d * Cc * 3 + w;
    for (int c = 0; c < Cc; ++c) s += xr[c] * twr[c * 3];
  }
  float ang = (float)l * expf(-9.210340371976184f * (float)(d & ~1) / (float)Dc);
  float pe  = (d & 1) ? cosf(ang) : sinf(ang);
  float v = s + pe;
  h[idx]  = v;
  hb[idx] = f2bf(v);
}

// sigma = h @ sig_w^T + sig_b, then sigma -> 3^(sigmoid(5*s)+1e-5) - 1
__global__ __launch_bounds__(256) void sigma_kernel(
    const float* __restrict__ h, const float* __restrict__ sw,
    const float* __restrict__ sb, float* __restrict__ out)
{
  int idx = blockIdx.x * 256 + threadIdx.x;        // (b*L, hd)
  int hd = idx & (Hc - 1);
  long long row = idx >> 3;
  const float* hr = h  + row * Dc;
  const float* wr = sw + (long long)hd * Dc;
  float s = sb[hd];
  for (int k = 0; k < Dc; ++k) s += hr[k] * wr[k];
  s = 1.f / (1.f + expf(-5.f * s)) + 1e-5f;
  s = expf(1.0986122886681098f * s) - 1.f;         // 3^s - 1
  out[idx] = s;
}

// in-place softmax over rows of 256; one wave per row; also bf16 copy
__global__ __launch_bounds__(256) void softmax_kernel(
    float* __restrict__ io, unsigned short* __restrict__ ob)
{
  long long row = (long long)blockIdx.x * 8 + (threadIdx.x >> 5);
  int lane = threadIdx.x & 31;
  float* p = io + row * Lc;
  unsigned short* q = ob + row * Lc;
  float v[8]; float mx = -3.4e38f;
  #pragma unroll
  for (int j = 0; j < 8; ++j) { v[j] = p[lane + j * 32]; mx = fmaxf(mx, v[j]); }
  #pragma unroll
  for (int o = 16; o > 0; o >>= 1) mx = fmaxf(mx, __shfl_xor(mx, o, 32));
  float s = 0.f;
  #pragma unroll
  for (int j = 0; j < 8; ++j) { v[j] = expf(v[j] - mx); s += v[j]; }
  #pragma unroll
  for (int o = 16; o > 0; o >>= 1) s += __shfl_xor(s, o, 32);
  float inv = 1.f / s;
  #pragma unroll
  for (int j = 0; j < 8; ++j) {
    float r = v[j] * inv;
    p[lane + j * 32] = r;
    q[lane + j * 32] = f2bf(r);
  }
}

// prior[b,h,l,s] = N(|l-s|; 0, sigma[b,l,h])
__global__ __launch_bounds__(256) void prior_kernel(
    const float* __restrict__ sig, float* __restrict__ out)
{
  long long idx = (long long)blockIdx.x * 256 + threadIdx.x;
  int s  = (int)(idx & 255);
  int l  = (int)((idx >> 8) & 255);
  int hd = (int)((idx >> 16) & 7);
  int b  = (int)(idx >> 19);
  float sg = sig[(((long long)b * Lc + l) << 3) + hd];
  float d  = fabsf((float)(l - s));
  out[idx] = 0.3989422804014327f / sg * expf(-(d * d) / (2.f * sg * sg));
}

// h = LayerNorm(h + add) ; writes fp32 and bf16. add may be null (plain LN).
__global__ __launch_bounds__(256) void add_ln_kernel(
    const float* __restrict__ add, float* __restrict__ h,
    unsigned short* __restrict__ hb,
    const float* __restrict__ g, const float* __restrict__ bta)
{
  __shared__ float red[256];
  long long base = (long long)blockIdx.x * Dc;
  int t = threadIdx.x;
  float x0 = h[base + t], x1 = h[base + t + 256];
  if (add) { x0 += add[base + t]; x1 += add[base + t + 256]; }
  red[t] = x0 + x1; __syncthreads();
  for (int o = 128; o > 0; o >>= 1) { if (t < o) red[t] += red[t + o]; __syncthreads(); }
  float mean = red[0] * (1.f / (float)Dc);
  __syncthreads();
  float d0 = x0 - mean, d1 = x1 - mean;
  red[t] = d0 * d0 + d1 * d1; __syncthreads();
  for (int o = 128; o > 0; o >>= 1) { if (t < o) red[t] += red[t + o]; __syncthreads(); }
  float inv = rsqrtf(red[0] * (1.f / (float)Dc) + 1e-5f);
  float y0 = d0 * inv * g[t]       + bta[t];
  float y1 = d1 * inv * g[t + 256] + bta[t + 256];
  h[base + t] = y0;           h[base + t + 256] = y1;
  hb[base + t] = f2bf(y0);    hb[base + t + 256] = f2bf(y1);
}

// z = h @ lat_w^T + lat_b (D -> 16, too skinny for WMMA)
__global__ __launch_bounds__(256) void latent_kernel(
    const float* __restrict__ h, const float* __restrict__ lw,
    const float* __restrict__ lb, float* __restrict__ outz)
{
  int idx = blockIdx.x * 256 + threadIdx.x;        // (b*L, zi)
  int zi = idx & (Zc - 1);
  long long row = idx >> 4;
  const float* hr = h  + row * Dc;
  const float* wr = lw + (long long)zi * Dc;
  float s = lb[zi];
  for (int k = 0; k < Dc; ++k) s += hr[k] * wr[k];
  outz[idx] = s;
}

__global__ __launch_bounds__(256) void dec1_kernel(
    const float* __restrict__ zv, const float* __restrict__ w,
    const float* __restrict__ b, float* __restrict__ hid)
{
  int idx = blockIdx.x * 256 + threadIdx.x;        // (b*L, d)
  int d = idx & (Dc - 1);
  long long row = idx >> 9;
  const float* zr = zv + row * Zc;
  const float* wr = w + (long long)d * Zc;
  float s = b[d];
  #pragma unroll
  for (int k = 0; k < Zc; ++k) s += zr[k] * wr[k];
  hid[idx] = fmaxf(s, 0.f);
}

__global__ __launch_bounds__(256) void recon_kernel(
    const float* __restrict__ hid, const float* __restrict__ w,
    const float* __restrict__ b, float* __restrict__ out)
{
  int idx = blockIdx.x * 256 + threadIdx.x;        // (b*L, c)
  int c = idx % Cc;
  long long row = idx / Cc;
  const float* hr = hid + row * Dc;
  const float* wr = w + (long long)c * Dc;
  float s = b[c];
  for (int k = 0; k < Dc; ++k) s += hr[k] * wr[k];
  out[idx] = s;
}

// ---------------------------------------------------------------------------
extern "C" void kernel_launch(void* const* d_in, const int* in_sizes, int n_in,
                              void* d_out, int out_size, void* d_ws, size_t ws_size,
                              hipStream_t stream) {
  const float* x       = (const float*)d_in[0];
  const float* tok_w   = (const float*)d_in[1];
  const float* qkv_w   = (const float*)d_in[2];
  const float* qkv_b   = (const float*)d_in[3];
  const float* sig_w   = (const float*)d_in[4];
  const float* sig_b   = (const float*)d_in[5];
  const float* out_w   = (const float*)d_in[6];
  const float* out_b   = (const float*)d_in[7];
  const float* conv1_w = (const float*)d_in[8];
  const float* conv1_b = (const float*)d_in[9];
  const float* conv2_w = (const float*)d_in[10];
  const float* conv2_b = (const float*)d_in[11];
  const float* ln1_g   = (const float*)d_in[12];
  const float* ln1_b   = (const float*)d_in[13];
  const float* ln2_g   = (const float*)d_in[14];
  const float* ln2_b   = (const float*)d_in[15];
  const float* lnf_g   = (const float*)d_in[16];
  const float* lnf_b   = (const float*)d_in[17];
  const float* lat_w   = (const float*)d_in[18];
  const float* lat_b   = (const float*)d_in[19];
  const float* dec1_w  = (const float*)d_in[20];
  const float* dec1_b  = (const float*)d_in[21];
  const float* dec2_w  = (const float*)d_in[22];
  const float* dec2_b  = (const float*)d_in[23];

  // ---- workspace carve-up (with lifetime-based reuse) ----
  char* ws = (char*)d_ws;
  size_t off = 0;
  auto take = [&](size_t bytes) -> char* {
    char* p = ws + off;
    off += (bytes + 255) & ~(size_t)255;
    return p;
  };
  unsigned short* wq   = (unsigned short*)take((size_t)9 * DDsz * 2);  // qkv_w bf16
  unsigned short* wo   = (unsigned short*)take((size_t)3 * DDsz * 2);  // out_w bf16
  unsigned short* wc1  = (unsigned short*)take((size_t)3 * DDsz * 2);  // conv1_w bf16
  unsigned short* wc2  = (unsigned short*)take((size_t)3 * DDsz * 2);  // conv2_w bf16
  float*          h    = (float*)take((size_t)BLD * 4);
  unsigned short* hbf  = (unsigned short*)take((size_t)BLD * 2);
  unsigned short* qbf  = (unsigned short*)take((size_t)BLD * 2);
  unsigned short* kbf  = (unsigned short*)take((size_t)BLD * 2);
  unsigned short* vbf  = (unsigned short*)take((size_t)BLD * 2);
  float*          sigp = (float*)take((size_t)Bc * Lc * Hc * 4);
  unsigned short* sbf  = (unsigned short*)take((size_t)BHLL * 2); // series bf16; reused as y1 bf16
  unsigned short* vagg = (unsigned short*)take((size_t)BLD * 2);  // attention output bf16
  float*          tmpf = (float*)take((size_t)BLD * 4);           // newh / y2 / hid

  float* fout      = (float*)d_out;
  float* outRecon  = fout;
  float* outSeries = fout + RECON;
  float* outPrior  = outSeries + (long long)NLc * BHLL;
  float* outZ      = outPrior  + (long long)NLc * BHLL;

  auto gemm = [&](const unsigned short* A, int lda,
                  const unsigned short* Bm, int ldb, int bKN,
                  const float* bias, float* oF, unsigned short* oB, int ldc,
                  int M, int N, int K, float alpha, int epi,
                  int nz, int zDiv,
                  long long sAo, long long sAi, long long sBo, long long sBi,
                  long long sCo, long long sCi) {
    dim3 g(M / 128, N / 64, nz);
    gemm_bf16_wmma<<<g, 256, 0, stream>>>(A, lda, Bm, ldb, bKN, bias, oF, oB, ldc,
                                          K, alpha, epi, zDiv,
                                          sAo, sAi, sBo, sBi, sCo, sCi);
  };

  // 1) weights -> bf16
  cvt_f32_bf16<<<(9 * DDsz) / 256, 256, 0, stream>>>(qkv_w,   wq,  9 * DDsz);
  cvt_f32_bf16<<<(3 * DDsz) / 256, 256, 0, stream>>>(out_w,   wo,  3 * DDsz);
  cvt_f32_bf16<<<(3 * DDsz) / 256, 256, 0, stream>>>(conv1_w, wc1, 3 * DDsz);
  cvt_f32_bf16<<<(3 * DDsz) / 256, 256, 0, stream>>>(conv2_w, wc2, 3 * DDsz);

  // 2) embedding
  embed_kernel<<<BLD / 256, 256, 0, stream>>>(x, tok_w, h, hbf);

  const long long LD = (long long)Lc * Dc;      // 131072
  const long long LL = (long long)Lc * Lc;      // 65536

  for (int i = 0; i < NLc; ++i) {
    // q,k,v projections (bf16 out only)
    unsigned short* qkvDst[3] = { qbf, kbf, vbf };
    for (int j = 0; j < 3; ++j)
      gemm(hbf, Dc, wq + (size_t)(i * 3 + j) * DDsz, Dc, 0,
           qkv_b + (size_t)(i * 3 + j) * Dc, nullptr, qkvDst[j], Dc,
           Bc * Lc, Dc, Dc, 1.f, 0, 1, 1, 0,0,0,0,0,0);

    // sigma head (with transform fused)
    sigma_kernel<<<(Bc * Lc * Hc) / 256, 256, 0, stream>>>(
        h, sig_w + (size_t)i * Hc * Dc, sig_b + (size_t)i * Hc, sigp);

    // scores = (Q K^T)/8 -> written straight into series slab
    float* serSlab = outSeries + (size_t)i * BHLL;
    gemm(qbf, Dc, kbf, Dc, 0, nullptr, serSlab, nullptr, Lc,
         Lc, Lc, DKc, 0.125f, 0, Bc * Hc, Hc,
         LD, 64, LD, 64, (long long)Hc * LL, LL);

    // softmax in place (+ bf16 copy for aggregation)
    softmax_kernel<<<(Bc * Hc * Lc) / 8, 256, 0, stream>>>(serSlab, sbf);

    // prior slab
    prior_kernel<<<BHLL / 256, 256, 0, stream>>>(sigp, outPrior + (size_t)i * BHLL);

    // V aggregation: series @ V  (B is K x N row-major -> bKN=1)
    gemm(sbf, Lc, vbf, Dc, 1, nullptr, nullptr, vagg, Dc,
         Lc, DKc, Lc, 1.f, 0, Bc * Hc, Hc,
         (long long)Hc * LL, LL, LD, 64, LD, 64);

    // output projection -> tmpf (fp32)
    gemm(vagg, Dc, wo + (size_t)i * DDsz, Dc, 0, out_b + (size_t)i * Dc,
         tmpf, nullptr, Dc, Bc * Lc, Dc, Dc, 1.f, 0, 1, 1, 0,0,0,0,0,0);

    // h = LN1(h + attn_out)
    add_ln_kernel<<<Bc * Lc, 256, 0, stream>>>(
        tmpf, h, hbf, ln1_g + (size_t)i * Dc, ln1_b + (size_t)i * Dc);

    // FFN: y1 = gelu(h @ W1^T + b1) -> bf16 (reuse sbf)
    gemm(hbf, Dc, wc1 + (size_t)i * DDsz, Dc, 0, conv1_b + (size_t)i * DFc,
         nullptr, sbf, DFc, Bc * Lc, DFc, Dc, 1.f, 1, 1, 1, 0,0,0,0,0,0);
    //      y2 = y1 @ W2^T + b2 -> fp32 (reuse tmpf)
    gemm(sbf, DFc, wc2 + (size_t)i * DDsz, DFc, 0, conv2_b + (size_t)i * Dc,
         tmpf, nullptr, Dc, Bc * Lc, Dc, DFc, 1.f, 0, 1, 1, 0,0,0,0,0,0);

    // h = LN2(h + y2)
    add_ln_kernel<<<Bc * Lc, 256, 0, stream>>>(
        tmpf, h, hbf, ln2_g + (size_t)i * Dc, ln2_b + (size_t)i * Dc);
  }

  // final LN
  add_ln_kernel<<<Bc * Lc, 256, 0, stream>>>(nullptr, h, hbf, lnf_g, lnf_b);

  // latent + decoder
  latent_kernel<<<(Bc * Lc * Zc) / 256, 256, 0, stream>>>(h, lat_w, lat_b, outZ);
  dec1_kernel<<<BLD / 256, 256, 0, stream>>>(outZ, dec1_w, dec1_b, tmpf);
  recon_kernel<<<RECON / 256, 256, 0, stream>>>(tmpf, dec2_w, dec2_b, outRecon);
}
